// SlowStrategicReasoner_18219251270373
// MI455X (gfx1250) — matrix operand order repaired and checked by reading, compiled, not run on hardware
//
#include <hip/hip_runtime.h>
#include <hip/hip_bf16.h>
#include <stdint.h>

// ---------------------------------------------------------------------------
// CDNA5 (gfx1250) implementation. wave32. WMMA bf16 16x16x32, f32 accumulate.
// bf16 everywhere in memory (weights pre-transposed+converted once), double-
// buffered async global->LDS tile staging (ASYNCcnt software pipeline),
// 4 WMMAs/wave/K-step, WMMA edge MLP.
// ---------------------------------------------------------------------------

typedef __attribute__((ext_vector_type(16))) __bf16 v16bf;
typedef __attribute__((ext_vector_type(8)))  __bf16 v8bf;
typedef __attribute__((ext_vector_type(8)))  float  v8f;

static __device__ __forceinline__ v16bf join16(v8bf lo, v8bf hi) {
  return __builtin_shufflevector(lo, hi, 0,1,2,3,4,5,6,7,8,9,10,11,12,13,14,15);
}

// LDS byte offset of a generic pointer to __shared__ memory.
static __device__ __forceinline__ uint32_t lds_off(void* p) {
  return (uint32_t)(uintptr_t)(__attribute__((address_space(3))) void*)p;
}

// GLOBAL_LOAD_ASYNC_TO_LDS_B128: copy 16B global -> LDS, tracked by ASYNCcnt.
static __device__ __forceinline__ void async_cp16(void* lds, const void* g) {
  asm volatile("global_load_async_to_lds_b128 %0, %1, off"
               :: "v"(lds_off(lds)),
                  "v"((unsigned long long)(uintptr_t)g)
               : "memory");
}
// Drain all async ops / all but the newest tile group (3 ops).
static __device__ __forceinline__ void async_wait0() {
  asm volatile("s_wait_asynccnt 0x0" ::: "memory");
}
static __device__ __forceinline__ void async_wait3() {
  asm volatile("s_wait_asynccnt 0x3" ::: "memory");
}

// ---------------------------------------------------------------------------
// GEMM: C[M,Nc](bf16) = act(A[M,K](bf16) @ WT[Nc,K](bf16,transposed) + bias)
// Block = 256 threads (8 waves as 2(M) x 4(N)); block tile 128x64; wave tile
// 64x16 -> 4 accumulators, 4 WMMAs per 32-wide K-step. Double-buffered LDS
// tiles filled by async b128 copies; s_wait_asynccnt 3 keeps the next tile in
// flight while the current one is consumed. M%128==0, K%32==0.
// ---------------------------------------------------------------------------
__global__ __launch_bounds__(256) void gemm_wmma_kernel(
    const __bf16* __restrict__ A, const __bf16* __restrict__ WT,
    const float* __restrict__ bias, __bf16* __restrict__ C,
    int M, int K, int Nc, int do_relu)
{
  __shared__ alignas(32) __bf16 As[2][128 * 32];  // [m][k], pitch 32 (64B/row)
  __shared__ alignas(32) __bf16 Bt[2][64 * 32];   // [n][k], pitch 32

  const int tid  = threadIdx.x;
  const int lane = tid & 31;
  const int wave = tid >> 5;            // 0..7
  const int mi   = wave >> 2;           // 0..1
  const int ni   = wave & 3;            // 0..3
  const int half = lane >> 4;           // 0 or 1
  const int l16  = lane & 15;

  const int n0 = blockIdx.x * 64;
  const int m0 = blockIdx.y * 128;

  // --- async staging assignments (loop-invariant source pointers)
  // A: thread covers 32B = 16 elems of row (tid>>1) at seg (tid&1)*16.
  const int a_row = tid >> 1, a_seg = (tid & 1) << 4;
  const __bf16* __restrict__ a_src = A + (size_t)(m0 + a_row) * (size_t)K + a_seg;
  // B: thread covers 16B = 8 elems of WT row n0+(tid>>2) at seg (tid&3)*8.
  const int b_row = tid >> 2, b_seg = (tid & 3) << 3;
  int b_n = n0 + b_row; if (b_n >= Nc) b_n = Nc - 1;   // clamp (Nc%64!=0)
  const __bf16* __restrict__ b_src = WT + (size_t)b_n * (size_t)K + b_seg;

  __bf16* a_dst[2] = { &As[0][a_row * 32 + a_seg], &As[1][a_row * 32 + a_seg] };
  __bf16* b_dst[2] = { &Bt[0][b_row * 32 + b_seg], &Bt[1][b_row * 32 + b_seg] };

  v8f acc0 = {}, acc1 = {}, acc2 = {}, acc3 = {};

  // prologue: issue tile 0 into buffer 0
  async_cp16(a_dst[0],     a_src);
  async_cp16(a_dst[0] + 8, a_src + 8);
  async_cp16(b_dst[0],     b_src);

  int buf = 0;
  for (int k0 = 0; k0 < K; k0 += 32, buf ^= 1) {
    const int nk = k0 + 32;
    if (nk < K) {
      // issue next tile into the other buffer, then wait for current tile
      // (in-order completion: <=3 outstanding leaves only the new group).
      const int nb = buf ^ 1;
      async_cp16(a_dst[nb],     a_src + nk);
      async_cp16(a_dst[nb] + 8, a_src + nk + 8);
      async_cp16(b_dst[nb],     b_src + nk);
      async_wait3();
    } else {
      async_wait0();
    }
    __syncthreads();   // all waves' slices of current tile visible

    // ---- fragments straight from LDS (per lane k = 8h+s, 16+8h+(s-8), ISA 7.12.2)
    const __bf16* __restrict__ bfp = &Bt[buf][(ni * 16 + l16) * 32 + half * 8];
    const __bf16* __restrict__ afp = &As[buf][(mi * 64 + l16) * 32 + half * 8];
    const v16bf bfrag = join16(*(const v8bf*)bfp, *(const v8bf*)(bfp + 16));   // 2x 16B aligned LDS loads
#pragma unroll
    for (int s = 0; s < 4; ++s) {
      const __bf16* ap = afp + s * 16 * 32;
      v16bf afrag = join16(*(const v8bf*)ap, *(const v8bf*)(ap + 16));
      v8f* accp = (s == 0) ? &acc0 : (s == 1) ? &acc1 : (s == 2) ? &acc2 : &acc3;
      *accp = __builtin_amdgcn_wmma_f32_16x16x32_bf16(
          false, afrag, false, bfrag, (short)0, *accp, false, false);
    }
    __syncthreads();   // done reading before this buffer is refilled (i+2)
  }

  // ---- epilogue: C VGPR r -> M = base + r + half*8, N = n0 + ni*16 + l16
  const int col = n0 + ni * 16 + l16;
  if (col < Nc) {
    const float bv = bias ? bias[col] : 0.0f;
#pragma unroll
    for (int s = 0; s < 4; ++s) {
      const v8f* accp = (s == 0) ? &acc0 : (s == 1) ? &acc1 : (s == 2) ? &acc2 : &acc3;
      const int rbase = m0 + mi * 64 + s * 16 + (half << 3);
#pragma unroll
      for (int r = 0; r < 8; ++r) {
        float v = (*accp)[r] + bv;
        if (do_relu) v = fmaxf(v, 0.0f);
        C[(size_t)(rbase + r) * (size_t)Nc + col] = (__bf16)v;
      }
    }
  }
}

// Weight prep: out[n*K + k] = (bf16) in[k*Nc + n]   (transpose + convert)
__global__ __launch_bounds__(256) void transpose_cvt_kernel(
    const float* __restrict__ in, __bf16* __restrict__ out, int K, int Nc)
{
  const int k = blockIdx.x * blockDim.x + threadIdx.x;
  const int n = blockIdx.y;
  if (k < K) out[(size_t)n * K + k] = (__bf16)in[(size_t)k * Nc + n];
}

// nodes(bf16)[r][d] = state[b, linspace-idx(i), d],  r = b*64+i
__global__ __launch_bounds__(256) void gather_cvt_kernel(
    const float* __restrict__ state, __bf16* __restrict__ nodes, int S, int D)
{
  const int d = blockIdx.x * blockDim.x + threadIdx.x;
  const int r = blockIdx.y;
  const int b = r >> 6, i = r & 63;
  const int idx = (int)((float)i * ((float)(S - 1) / 63.0f));
  nodes[(size_t)r * D + d] = (__bf16)state[((size_t)b * S + idx) * D + d];
}

// Row LayerNorm (bf16 in), optional nan_to_num; bf16 or f32 out.
__global__ __launch_bounds__(128) void layernorm_kernel(
    const __bf16* __restrict__ in, const float* __restrict__ g,
    const float* __restrict__ be, void* __restrict__ out,
    int width, int nan2num, int out_f32)
{
  const int row = blockIdx.x;
  const __bf16* __restrict__ x = in + (size_t)row * width;
  float s = 0.0f, s2 = 0.0f;
  for (int c = threadIdx.x; c < width; c += blockDim.x) {
    const float v = (float)x[c];
    s += v; s2 += v * v;
  }
  __shared__ float rs[8], rs2[8];
#pragma unroll
  for (int off = 16; off > 0; off >>= 1) {
    s  += __shfl_down(s,  off, 32);
    s2 += __shfl_down(s2, off, 32);
  }
  const int w = threadIdx.x >> 5, ln = threadIdx.x & 31;
  if (ln == 0) { rs[w] = s; rs2[w] = s2; }
  __syncthreads();
  if (threadIdx.x == 0) {
    float ts = 0.0f, ts2 = 0.0f;
    const int nw = blockDim.x >> 5;
    for (int i = 0; i < nw; ++i) { ts += rs[i]; ts2 += rs2[i]; }
    rs[0] = ts; rs2[0] = ts2;
  }
  __syncthreads();
  const float mean = rs[0] / (float)width;
  const float var  = rs2[0] / (float)width - mean * mean;
  const float inv  = rsqrtf(var + 1e-5f);
  for (int c = threadIdx.x; c < width; c += blockDim.x) {
    float v = ((float)x[c] - mean) * inv * g[c] + be[c];
    if (nan2num) {
      if (isnan(v)) v = 0.0f;
      else if (isinf(v)) v = (v > 0.0f) ? 3.4028234663852886e38f : -3.4028234663852886e38f;
    }
    if (out_f32) ((float*)out)[(size_t)row * width + c] = v;
    else         ((__bf16*)out)[(size_t)row * width + c] = (__bf16)v;
  }
}

// ---------------------------------------------------------------------------
// Fused edge + strength kernel, WMMA-based. Block = one (b,i), 64 threads
// (2 waves). h(64x64) @ W_e2(64x32): 2 waves x 4 M-subtiles x 2 K-steps =
// 16 WMMAs; epilogue relu -> *W_e3 partials in LDS -> 32-wide row sum.
// ---------------------------------------------------------------------------
__global__ __launch_bounds__(64) void edge_kernel(
    const __bf16* __restrict__ hi, const __bf16* __restrict__ hj,
    const __bf16* __restrict__ si, const __bf16* __restrict__ sj,
    const float* __restrict__ b_e1, const __bf16* __restrict__ W_e2T,
    const float* __restrict__ b_e2, const float* __restrict__ W_e3,
    const float* __restrict__ b_e3, const float* __restrict__ b_s1,
    const float* __restrict__ W_s2, const float* __restrict__ b_s2,
    float* __restrict__ adj, float* __restrict__ strength)
{
  const int bi = blockIdx.x;        // = b*64 + i
  const int b  = bi >> 6;
  const int i  = bi & 63;
  const int tid  = threadIdx.x;     // 0..63
  const int wave = tid >> 5;        // N columns wave*16..+15
  const int lane = tid & 31;
  const int half = lane >> 4;
  const int l16  = lane & 15;

  __shared__ alignas(32) __bf16 Hs[64 * 64];   // h rows [j][k], pitch 64
  __shared__ alignas(32) __bf16 Wt[32 * 64];   // W_e2^T [m][k], pitch 64
  __shared__ float Tm[64 * 33];                // relu(t)*w3 partials, padded
  __shared__ float his[64], be1s[64];
  __shared__ float sis[32], bs1s[32], ws2s[32], w3s[32], b2s[32];

  // async-copy W_e2^T (32x64 bf16 = 4KB, layout identical to Wt): 64B/thread
  {
    char* dst = (char*)Wt + tid * 64;
    const char* src = (const char*)W_e2T + tid * 64;
    async_cp16(dst,      src);
    async_cp16(dst + 16, src + 16);
    async_cp16(dst + 32, src + 32);
    async_cp16(dst + 48, src + 48);
  }

  his[tid]  = (float)hi[(size_t)bi * 64 + tid];
  be1s[tid] = b_e1[tid];
  if (tid < 32) {
    sis[tid]  = (float)si[(size_t)bi * 32 + tid];
    b2s[tid]  = b_e2[tid];
    w3s[tid]  = W_e3[tid];
    bs1s[tid] = b_s1[tid];
    ws2s[tid] = W_s2[tid];
  }
  __syncthreads();

  // Build h row j = tid: relu(hi[b,i] + hj[b,j] + b_e1), bf16 into LDS.
  {
    const __bf16* __restrict__ hjrow = hj + (size_t)(b * 64 + tid) * 64;
    __bf16* dst = &Hs[tid * 64];
#pragma unroll
    for (int k = 0; k < 64; ++k)
      dst[k] = (__bf16)fmaxf(his[k] + (float)hjrow[k] + be1s[k], 0.0f);
  }
  async_wait0();
  __syncthreads();

  // GEMM h @ W_e2 (M=64 rows j, K=64, N=32).
  v8f acc0 = {}, acc1 = {}, acc2 = {}, acc3 = {};
#pragma unroll
  for (int kk = 0; kk < 2; ++kk) {
    const v16bf bfrag = join16(*(const v8bf*)&Wt[(wave * 16 + l16) * 64 + kk * 32 + half * 8], *(const v8bf*)&Wt[(wave * 16 + l16) * 64 + kk * 32 + half * 8 + 16]);
#pragma unroll
    for (int s = 0; s < 4; ++s) {
      const __bf16* ap = &Hs[(s * 16 + l16) * 64 + kk * 32 + half * 8];
      v16bf afrag = join16(*(const v8bf*)ap, *(const v8bf*)(ap + 16));
      v8f* accp = (s == 0) ? &acc0 : (s == 1) ? &acc1 : (s == 2) ? &acc2 : &acc3;
      *accp = __builtin_amdgcn_wmma_f32_16x16x32_bf16(
          false, afrag, false, bfrag, (short)0, *accp, false, false);
    }
  }

  // Epilogue: Tm[j][m] = relu(t + b_e2[m]) * W_e3[m]
  {
    const int m = wave * 16 + l16;
    const float bm = b2s[m], wm = w3s[m];
#pragma unroll
    for (int s = 0; s < 4; ++s) {
      const v8f* accp = (s == 0) ? &acc0 : (s == 1) ? &acc1 : (s == 2) ? &acc2 : &acc3;
#pragma unroll
      for (int r = 0; r < 8; ++r) {
        const int row = s * 16 + r + (half << 3);
        Tm[row * 33 + m] = fmaxf((*accp)[r] + bm, 0.0f) * wm;
      }
    }
  }
  __syncthreads();

  // Per-pair scalar tail: thread j = tid.
  float a = b_e3[0];
#pragma unroll
  for (int m = 0; m < 32; ++m) a += Tm[tid * 33 + m];
  float adjv = 1.0f / (1.0f + expf(-a));

  float sa = b_s2[0];
  {
    const __bf16* __restrict__ sjrow = sj + (size_t)(b * 64 + tid) * 32;
#pragma unroll
    for (int k = 0; k < 32; ++k)
      sa += fmaxf(sis[k] + (float)sjrow[k] + bs1s[k], 0.0f) * ws2s[k];
  }
  float stv = tanhf(sa);

  if (i == tid) { adjv = 0.0f; stv = 0.0f; }   // off-diagonal mask
  if (!(adjv > 0.3f)) adjv = 0.0f;             // threshold

  adj[(size_t)bi * 64 + tid]      = adjv;
  strength[(size_t)bi * 64 + tid] = stv;
}

// gmean[b, c] = mean_i x[b, i, c]
__global__ __launch_bounds__(256) void gmean_kernel(
    const __bf16* __restrict__ x, __bf16* __restrict__ gmean)
{
  const int b = blockIdx.x, c = threadIdx.x;   // c in [0,256)
  float s = 0.0f;
#pragma unroll 8
  for (int i = 0; i < 64; ++i) s += (float)x[((size_t)(b * 64 + i)) * 256 + c];
  gmean[(size_t)b * 256 + c] = (__bf16)(s * (1.0f / 64.0f));
}

// cg[b] = concat(causal[b] (256), goals[b] (128))
__global__ __launch_bounds__(384) void concat_kernel(
    const __bf16* __restrict__ causal, const __bf16* __restrict__ goals,
    __bf16* __restrict__ cg)
{
  const int b = blockIdx.x, c = threadIdx.x;
  cg[(size_t)b * 384 + c] =
      (c < 256) ? causal[(size_t)b * 256 + c] : goals[(size_t)b * 128 + (c - 256)];
}

// ---------------------------------------------------------------------------
extern "C" void kernel_launch(void* const* d_in, const int* in_sizes, int n_in,
                              void* d_out, int out_size, void* d_ws, size_t ws_size,
                              hipStream_t stream) {
  (void)in_sizes; (void)n_in; (void)out_size; (void)ws_size;

  const float* state  = (const float*)d_in[0];
  const float* W_n1   = (const float*)d_in[1];
  const float* b_n1   = (const float*)d_in[2];
  const float* W_n2   = (const float*)d_in[3];
  const float* b_n2   = (const float*)d_in[4];
  const float* g_n    = (const float*)d_in[5];
  const float* be_n   = (const float*)d_in[6];
  const float* W_e1   = (const float*)d_in[7];
  const float* b_e1   = (const float*)d_in[8];
  const float* W_e2   = (const float*)d_in[9];
  const float* b_e2   = (const float*)d_in[10];
  const float* W_e3   = (const float*)d_in[11];
  const float* b_e3   = (const float*)d_in[12];
  const float* W_s1   = (const float*)d_in[13];
  const float* b_s1   = (const float*)d_in[14];
  const float* W_s2   = (const float*)d_in[15];
  const float* b_s2   = (const float*)d_in[16];
  const float* W_p1   = (const float*)d_in[17];
  const float* b_p1   = (const float*)d_in[18];
  const float* W_p2   = (const float*)d_in[19];
  const float* b_p2   = (const float*)d_in[20];
  const float* W_g1   = (const float*)d_in[21];
  const float* b_g1   = (const float*)d_in[22];
  const float* W_g2   = (const float*)d_in[23];
  const float* b_g2   = (const float*)d_in[24];
  const float* g_g    = (const float*)d_in[25];
  const float* be_g   = (const float*)d_in[26];
  const float* W_gg1  = (const float*)d_in[27];
  const float* b_gg1  = (const float*)d_in[28];
  const float* W_gg2  = (const float*)d_in[29];
  const float* b_gg2  = (const float*)d_in[30];
  const float* g_goal = (const float*)d_in[31];
  const float* be_goal= (const float*)d_in[32];
  // 33..36: W_pr1/b_pr1/W_pr2/b_pr2 -> priorities are dead code in reference
  const float* W_h1   = (const float*)d_in[37];
  const float* b_h1   = (const float*)d_in[38];
  const float* W_h2   = (const float*)d_in[39];
  const float* b_h2   = (const float*)d_in[40];
  const float* g_h    = (const float*)d_in[41];
  const float* be_h   = (const float*)d_in[42];

  float* out       = (float*)d_out;
  float* strat_out = out;                       // 128*256
  float* adj_out   = out + 128 * 256;           // 128*64*64
  float* str_out   = adj_out + 128 * 64 * 64;   // 128*64*64

  // workspace bump allocator (256B aligned)
  uintptr_t p = (uintptr_t)d_ws;
  auto alloc = [&](size_t bytes) -> void* {
    uintptr_t q = (p + 255) & ~(uintptr_t)255;
    p = q + bytes;
    return (void*)q;
  };
  // bf16 transposed weights
  __bf16* Wn1T  = (__bf16*)alloc((size_t)1024 * 2048 * 2);
  __bf16* Wn2T  = (__bf16*)alloc((size_t)512 * 1024 * 2);
  __bf16* We1aT = (__bf16*)alloc((size_t)64 * 512 * 2);
  __bf16* We1bT = (__bf16*)alloc((size_t)64 * 512 * 2);
  __bf16* Ws1aT = (__bf16*)alloc((size_t)32 * 512 * 2);
  __bf16* Ws1bT = (__bf16*)alloc((size_t)32 * 512 * 2);
  __bf16* Wp1T  = (__bf16*)alloc((size_t)256 * 512 * 2);
  __bf16* Wp2T  = (__bf16*)alloc((size_t)256 * 256 * 2);
  __bf16* Wg1T  = (__bf16*)alloc((size_t)256 * 256 * 2);
  __bf16* Wg2T  = (__bf16*)alloc((size_t)256 * 256 * 2);
  __bf16* Wgg1T = (__bf16*)alloc((size_t)512 * 256 * 2);
  __bf16* Wgg2T = (__bf16*)alloc((size_t)128 * 512 * 2);
  __bf16* Wh1T  = (__bf16*)alloc((size_t)256 * 384 * 2);
  __bf16* Wh2T  = (__bf16*)alloc((size_t)256 * 256 * 2);
  __bf16* We2T  = (__bf16*)alloc((size_t)32 * 64 * 2);
  // bf16 activations
  __bf16* nodes = (__bf16*)alloc((size_t)8192 * 2048 * 2);
  __bf16* h1    = (__bf16*)alloc((size_t)8192 * 1024 * 2);
  __bf16* nf    = (__bf16*)alloc((size_t)8192 * 512 * 2);
  __bf16* hib   = (__bf16*)alloc((size_t)8192 * 64 * 2);
  __bf16* hjb   = (__bf16*)alloc((size_t)8192 * 64 * 2);
  __bf16* sib   = (__bf16*)alloc((size_t)8192 * 32 * 2);
  __bf16* sjb   = (__bf16*)alloc((size_t)8192 * 32 * 2);
  __bf16* x1    = (__bf16*)alloc((size_t)8192 * 256 * 2);
  __bf16* x2    = (__bf16*)alloc((size_t)8192 * 256 * 2);
  __bf16* gmean = (__bf16*)alloc((size_t)128 * 256 * 2);
  __bf16* c1    = (__bf16*)alloc((size_t)128 * 256 * 2);
  __bf16* causal= (__bf16*)alloc((size_t)128 * 256 * 2);
  __bf16* gg1   = (__bf16*)alloc((size_t)128 * 512 * 2);
  __bf16* goals = (__bf16*)alloc((size_t)128 * 128 * 2);
  __bf16* cg    = (__bf16*)alloc((size_t)128 * 384 * 2);
  __bf16* hb    = (__bf16*)alloc((size_t)128 * 256 * 2);
  __bf16* sp    = (__bf16*)alloc((size_t)128 * 256 * 2);

  auto tcvt = [&](const float* in, __bf16* o, int K, int Nc) {
    dim3 g((K + 255) / 256, Nc);
    transpose_cvt_kernel<<<g, 256, 0, stream>>>(in, o, K, Nc);
  };
  auto gemm = [&](const __bf16* A, const __bf16* WT, const float* bias,
                  __bf16* C, int M, int K, int Nc, int relu) {
    dim3 grid((Nc + 63) / 64, M / 128);
    gemm_wmma_kernel<<<grid, dim3(256), 0, stream>>>(A, WT, bias, C, M, K, Nc, relu);
  };

  // ---- one-shot prep: weights -> bf16 transposed; nodes gather -> bf16
  tcvt(W_n1, Wn1T, 2048, 1024);
  tcvt(W_n2, Wn2T, 1024, 512);
  tcvt(W_e1,            We1aT, 512, 64);
  tcvt(W_e1 + 512 * 64, We1bT, 512, 64);
  tcvt(W_s1,            Ws1aT, 512, 32);
  tcvt(W_s1 + 512 * 32, Ws1bT, 512, 32);
  tcvt(W_p1, Wp1T, 512, 256);
  tcvt(W_p2, Wp2T, 256, 256);
  tcvt(W_g1, Wg1T, 256, 256);
  tcvt(W_g2, Wg2T, 256, 256);
  tcvt(W_gg1, Wgg1T, 256, 512);
  tcvt(W_gg2, Wgg2T, 512, 128);
  tcvt(W_h1, Wh1T, 384, 256);
  tcvt(W_h2, Wh2T, 256, 256);
  tcvt(W_e2, We2T, 64, 32);
  gather_cvt_kernel<<<dim3(8, 8192), 256, 0, stream>>>(state, nodes, 128, 2048);

  // ---- nf = nan_to_num(ln(relu(nodes @ W_n1 + b_n1) @ W_n2 + b_n2))
  gemm(nodes, Wn1T, b_n1, h1, 8192, 2048, 1024, 1);
  gemm(h1, Wn2T, b_n2, nf, 8192, 1024, 512, 0);
  layernorm_kernel<<<8192, 128, 0, stream>>>(nf, g_n, be_n, nf, 512, 1, 0);

  // ---- edge / strength projections + pairwise MLP
  gemm(nf, We1aT, nullptr, hib, 8192, 512, 64, 0);
  gemm(nf, We1bT, nullptr, hjb, 8192, 512, 64, 0);
  gemm(nf, Ws1aT, nullptr, sib, 8192, 512, 32, 0);
  gemm(nf, Ws1bT, nullptr, sjb, 8192, 512, 32, 0);
  edge_kernel<<<8192, 64, 0, stream>>>(hib, hjb, sib, sjb, b_e1, We2T, b_e2,
                                       W_e3, b_e3, b_s1, W_s2, b_s2,
                                       adj_out, str_out);

  // ---- pooled path
  gemm(nf, Wp1T, b_p1, x1, 8192, 512, 256, 1);
  gemm(x1, Wp2T, b_p2, x2, 8192, 256, 256, 0);
  gmean_kernel<<<128, 256, 0, stream>>>(x2, gmean);

  gemm(gmean, Wg1T, b_g1, c1, 128, 256, 256, 1);
  gemm(c1, Wg2T, b_g2, causal, 128, 256, 256, 0);
  layernorm_kernel<<<128, 128, 0, stream>>>(causal, g_g, be_g, causal, 256, 0, 0);

  gemm(causal, Wgg1T, b_gg1, gg1, 128, 256, 512, 1);
  gemm(gg1, Wgg2T, b_gg2, goals, 128, 512, 128, 0);
  layernorm_kernel<<<128, 128, 0, stream>>>(goals, g_goal, be_goal, goals, 128, 1, 0);

  concat_kernel<<<128, 384, 0, stream>>>(causal, goals, cg);
  gemm(cg, Wh1T, b_h1, hb, 128, 384, 256, 1);
  gemm(hb, Wh2T, b_h2, sp, 128, 256, 256, 0);
  layernorm_kernel<<<128, 128, 0, stream>>>(sp, g_h, be_h, strat_out, 256, 0, 1);
}
